// PPIModel_24859270709495
// MI455X (gfx1250) — compile-verified, matrix operand order checked by reading, not run on hardware
//
#include <hip/hip_runtime.h>

#define HD 64
#define MT 5          // node tiles (of 16 rows) per wave: 3125 tiles = 625 waves * 5

typedef __attribute__((ext_vector_type(16))) _Float16 v16h;
typedef __attribute__((ext_vector_type(8)))  float    v8f;

// Packed weight layout per layer (f16, WMMA B per-lane order):
//   wp[((t*2 + ks)*32 + lane)*16 + j]
// t = 0..11 output n-tile (0-3: V[l,0], 4-7: V[l,1], 8-11: loop_w[l])
// ks = K-step (0: K 0..31, 1: K 32..63)
// lane: col = lane&15, half = lane>>4
// j<8:  K = ks*32 + half*8 + j ;  j>=8: K = ks*32 + 16 + half*8 + (j-8)
#define WPACK_PER_LAYER (12 * 2 * 32 * 16)   // 12288 f16 = 24 KB

// ---------------------------------------------------------------------------
// One-time weight repack: fp32 row-major -> f16 WMMA-B lane layout.
// ---------------------------------------------------------------------------
__global__ __launch_bounds__(256) void pack_weights(
    const float* __restrict__ V,       // [2,2,64,64]
    const float* __restrict__ loop_w,  // [2,64,64]
    _Float16* __restrict__ wp)         // [2, WPACK_PER_LAYER]
{
    const int id = blockIdx.x * blockDim.x + threadIdx.x;
    if (id >= 2 * WPACK_PER_LAYER) return;

    const int l    = id / WPACK_PER_LAYER;
    const int rem  = id % WPACK_PER_LAYER;
    const int j    = rem & 15;
    const int lane = (rem >> 4) & 31;
    const int ks   = (rem >> 9) & 1;
    const int t    = rem >> 10;

    const int half = lane >> 4;
    const int idx  = lane & 15;
    const int K    = ks * 32 + ((j < 8) ? (half * 8 + j) : (16 + half * 8 + (j - 8)));

    const float* W;
    int colbase;
    if (t < 4)      { W = V + (size_t)l * 2 * 64 * 64;            colbase = t << 4; }
    else if (t < 8) { W = V + (size_t)l * 2 * 64 * 64 + 64 * 64;  colbase = (t - 4) << 4; }
    else            { W = loop_w + (size_t)l * 64 * 64;           colbase = (t - 8) << 4; }

    wp[id] = (_Float16)W[K * HD + colbase + idx];
}

// ---------------------------------------------------------------------------
// Fused per-layer dense kernel: one wave handles MT=5 node tiles (80 rows).
//   proj[n, 0:64]   = act(x[n]) @ V[l,0]
//   proj[n, 64:128] = act(x[n]) @ V[l,1]
//   h[n, :]         = act(x[n]) @ loop_w[l] + bias[l]   (edge kernel adds on top)
// B tiles: one contiguous v16h load (2x global_load_b128), reused 5x.
// ---------------------------------------------------------------------------
__global__ __launch_bounds__(32) void rgcn_layer_gemm(
    const float* __restrict__ x,        // [N,64]
    const _Float16* __restrict__ wp,    // packed weights for this layer
    const float* __restrict__ bias,     // [64]
    float* __restrict__ proj,           // [N,128]
    float* __restrict__ h,              // [N,64]
    int relu_in)
{
    const int lane  = threadIdx.x & 31;
    const int half  = lane >> 4;
    const int idx   = lane & 15;
    const int tile0 = blockIdx.x * MT;

    // ---- A tiles (fp32 -> f16, ReLU folded): contiguous float4 loads
    v16h a[MT][2];
#pragma unroll
    for (int m = 0; m < MT; ++m) {
        const float* xr = x + (size_t)((tile0 + m) * 16 + idx) * HD;
#pragma unroll
        for (int g = 0; g < 4; ++g) {          // K groups 0,16,32,48
            float4 q0 = *(const float4*)(xr + g * 16 + half * 8);
            float4 q1 = *(const float4*)(xr + g * 16 + half * 8 + 4);
            if (relu_in) {
                q0.x = fmaxf(q0.x, 0.f); q0.y = fmaxf(q0.y, 0.f);
                q0.z = fmaxf(q0.z, 0.f); q0.w = fmaxf(q0.w, 0.f);
                q1.x = fmaxf(q1.x, 0.f); q1.y = fmaxf(q1.y, 0.f);
                q1.z = fmaxf(q1.z, 0.f); q1.w = fmaxf(q1.w, 0.f);
            }
            const int ks = g >> 1;             // 0,0,1,1
            const int jb = (g & 1) * 8;        // 0,8,0,8
            a[m][ks][jb + 0] = (_Float16)q0.x; a[m][ks][jb + 1] = (_Float16)q0.y;
            a[m][ks][jb + 2] = (_Float16)q0.z; a[m][ks][jb + 3] = (_Float16)q0.w;
            a[m][ks][jb + 4] = (_Float16)q1.x; a[m][ks][jb + 5] = (_Float16)q1.y;
            a[m][ks][jb + 6] = (_Float16)q1.z; a[m][ks][jb + 7] = (_Float16)q1.w;
        }
    }

    // ---- 12 output n-tiles, B loaded once and reused across MT node tiles
#pragma unroll
    for (int t = 0; t < 12; ++t) {
        const v16h b0 = *(const v16h*)(wp + ((t * 2 + 0) * 32 + lane) * 16);
        const v16h b1 = *(const v16h*)(wp + ((t * 2 + 1) * 32 + lane) * 16);
        const float cinit = (t >= 8) ? bias[((t - 8) << 4) + idx] : 0.f;

#pragma unroll
        for (int m = 0; m < MT; ++m) {
            v8f c;
#pragma unroll
            for (int v = 0; v < 8; ++v) c[v] = cinit;

            c = __builtin_amdgcn_wmma_f32_16x16x32_f16(false, a[m][0], false, b0,
                                                       (short)0, c, false, false);
            c = __builtin_amdgcn_wmma_f32_16x16x32_f16(false, a[m][1], false, b1,
                                                       (short)0, c, false, false);

            const int row0 = (tile0 + m) * 16 + 8 * half;    // D: row = v + 8*half
            if (t < 8) {
#pragma unroll
                for (int v = 0; v < 8; ++v)
                    proj[(size_t)(row0 + v) * 128 + (t << 4) + idx] = c[v];
            } else {
#pragma unroll
                for (int v = 0; v < 8; ++v)
                    h[(size_t)(row0 + v) * HD + ((t - 8) << 4) + idx] = c[v];
            }
        }
    }
}

// ---------------------------------------------------------------------------
// Edge message + scatter-add. One wave per edge; each lane owns 2 channels.
// proj (25.6 MB) and h (12.8 MB) are L2-resident (192 MB L2) -> gather/scatter
// stays on-chip; HBM only streams the edge lists.
// ---------------------------------------------------------------------------
__global__ __launch_bounds__(256) void rgcn_edge(
    const float* __restrict__ proj,    // [N,128]
    const float* __restrict__ compL,   // [8,2] coefficients for this layer
    const int*   __restrict__ src,
    const int*   __restrict__ dst,
    const int*   __restrict__ etype,
    float* __restrict__ h,             // [N,64] accumulate target
    int E)
{
    const int e    = blockIdx.x * (blockDim.x >> 5) + (threadIdx.x >> 5);
    const int lane = threadIdx.x & 31;
    if (e >= E) return;

    const int s = src[e];
    const int d = dst[e];
    const int r = etype[e];
    const float c0 = compL[r * 2 + 0];
    const float c1 = compL[r * 2 + 1];

    const float2* p  = (const float2*)(proj + (size_t)s * 128);
    const float2 p0 = p[lane];         // basis 0, channels 2*lane, 2*lane+1
    const float2 p1 = p[32 + lane];    // basis 1

    const float mx = c0 * p0.x + c1 * p1.x;
    const float my = c0 * p0.y + c1 * p1.y;

    float* out = h + (size_t)d * HD + 2 * lane;
    atomicAdd(out,     mx);
    atomicAdd(out + 1, my);
}

// ---------------------------------------------------------------------------
// Readout: dot(x_final, fc_w) reduced into a single float, then sigmoid.
// ---------------------------------------------------------------------------
__global__ __launch_bounds__(256) void fc_dot(
    const float* __restrict__ xf, const float* __restrict__ w,
    float* __restrict__ acc, int n)
{
    float s = 0.f;
    const int stride = gridDim.x * blockDim.x;
    for (int i = blockIdx.x * blockDim.x + threadIdx.x; i < n; i += stride)
        s += xf[i] * w[i];
#pragma unroll
    for (int off = 16; off > 0; off >>= 1)
        s += __shfl_down(s, off, 32);
    if ((threadIdx.x & 31) == 0)
        atomicAdd(acc, s);
}

__global__ void fc_finish(const float* __restrict__ acc,
                          const float* __restrict__ fc_b,
                          float* __restrict__ out)
{
    const float v = acc[0] + fc_b[0];
    out[0] = 1.0f / (1.0f + __expf(-v));
}

// ---------------------------------------------------------------------------
extern "C" void kernel_launch(void* const* d_in, const int* in_sizes, int n_in,
                              void* d_out, int out_size, void* d_ws, size_t ws_size,
                              hipStream_t stream) {
    const float* features   = (const float*)d_in[0];  // [N,64]
    const float* V          = (const float*)d_in[1];  // [2,2,64,64]
    const float* comp       = (const float*)d_in[2];  // [2,8,2]
    const float* loop_w     = (const float*)d_in[3];  // [2,64,64]
    const float* layer_bias = (const float*)d_in[4];  // [2,64]
    const float* fc_w       = (const float*)d_in[5];  // [N*64]
    const float* fc_b       = (const float*)d_in[6];  // [1]
    const int*   src        = (const int*)d_in[7];
    const int*   dst        = (const int*)d_in[8];
    const int*   etype      = (const int*)d_in[9];

    const int N = in_sizes[0] / HD;   // 50000 (multiple of 80)
    const int E = in_sizes[7];        // 800000

    // workspace layout (wpack first for 32B alignment of v16h loads)
    _Float16* wpack = (_Float16*)d_ws;                       // 2 * 12288 f16 = 48 KB
    float* proj = (float*)(wpack + 2 * WPACK_PER_LAYER);     // N*128
    float* h0   = proj + (size_t)N * 128;                    // N*64
    float* h1   = h0   + (size_t)N * HD;                     // N*64
    float* acc  = h1   + (size_t)N * HD;                     // 1

    hipMemsetAsync(acc, 0, sizeof(float), stream);

    pack_weights<<<(2 * WPACK_PER_LAYER + 255) / 256, 256, 0, stream>>>(
        V, loop_w, wpack);

    const int nblocks = N / (16 * MT);          // 625
    const int ewaves  = (E + 7) / 8;

    // ---- layer 0 (input = features, no relu on load)
    rgcn_layer_gemm<<<nblocks, 32, 0, stream>>>(
        features, wpack, layer_bias, proj, h0, 0);
    rgcn_edge<<<ewaves, 256, 0, stream>>>(
        proj, comp, src, dst, etype, h0, E);

    // ---- layer 1 (input = h0, relu folded into A-tile load)
    rgcn_layer_gemm<<<nblocks, 32, 0, stream>>>(
        h0, wpack + WPACK_PER_LAYER, layer_bias + HD, proj, h1, 1);
    rgcn_edge<<<ewaves, 256, 0, stream>>>(
        proj, comp + 16, src, dst, etype, h1, E);

    // ---- readout
    fc_dot<<<1024, 256, 0, stream>>>(h1, fc_w, acc, N * HD);
    fc_finish<<<1, 1, 0, stream>>>(acc, fc_b, (float*)d_out);
}
